// TopKSAE_41583873360699
// MI455X (gfx1250) — matrix operand order repaired and checked by reading, not compiled, not run on hardware
//
#include <hip/hip_runtime.h>
#include <stdint.h>

// ---------------- problem constants (from reference) ----------------
#define B_ROWS 2048
#define D_IN   1024
#define N_FEAT 65536
#define TOPK   32

// ---------------- tiling ----------------
#define ROWS_PER_WG 64                    // 4 row subtiles of 16
#define COLS_PER_WG 256                   // 2 col groups of 128 (8 WMMA tiles each)
#define NSLICE (N_FEAT / COLS_PER_WG)     // 256 column slices
#define NROWT  (B_ROWS / ROWS_PER_WG)     // 32 row tiles
#define KCHUNK 32                         // WMMA bf16 K per instruction
#define NKC    (D_IN / KCHUNK)            // 32 k-chunks

// LDS strides padded so the dword stride is odd -> conflict-free on 64 banks
#define AS_STRIDE (D_IN + 2)              // 1026 bf16 = 513 dwords (odd)
#define ES_STRIDE (KCHUNK + 2)            // 34 bf16   = 17 dwords (odd)

#define AS_BYTES (ROWS_PER_WG * AS_STRIDE * 2)   // 131,328
#define ES_BYTES (COLS_PER_WG * ES_STRIDE * 2)   //  17,408
#define SMEM_BYTES (AS_BYTES + ES_BYTES)         // 148,736 (< 160 KB CU-mode cap)

typedef __attribute__((ext_vector_type(16))) __bf16 v16bf;
typedef __attribute__((ext_vector_type(8)))  float  v8f;

static __device__ __forceinline__ uint32_t f32_to_bf16_bits(float f) {
  uint32_t u = __float_as_uint(f);
  u += 0x7FFFu + ((u >> 16) & 1u);        // round-to-nearest-even
  return u >> 16;
}

// =====================================================================
// Kernel 1: fused (x - pre_b) @ enc  +  per-row streaming top-32
// =====================================================================
__global__ __launch_bounds__(256)
void sae_encode_topk(const float* __restrict__ x,
                     const float* __restrict__ pre_b,
                     const float* __restrict__ enc,
                     float* __restrict__ candV,
                     int*   __restrict__ candI)
{
  __shared__ __align__(16) unsigned char smem[SMEM_BYTES];
  uint16_t* As = (uint16_t*)smem;                    // bf16 A tile [64][1026]
  uint16_t* Es = (uint16_t*)(smem + AS_BYTES);       // bf16 enc chunk [n][k]
  // Epilogue buffers alias the A tile (dead after the k-loop):
  float* stageV = (float*)smem;                      // [8][16][16]  8 KB
  float* twv    = (float*)(smem + 8192);             // [8][16][32] 16 KB
  int*   twi    = (int*)  (smem + 8192 + 16384);     // [8][16][32] 16 KB

  const int tid  = threadIdx.x;
  const int lane = tid & 31;
  const int w    = tid >> 5;            // wave 0..7
  const int rowt  = blockIdx.x;         // FAST dim: all row tiles share hot slices in L2
  const int slice = blockIdx.y;         // 0..255
  const int rowbase = rowt * ROWS_PER_WG;
  const int colbase = slice * COLS_PER_WG;
  const int sr = w >> 1;                // row subtile 0..3 (16 rows each)
  const int g  = w & 1;                 // col group 0..1  (128 cols each)

  // ---- stage A = (x - pre_b), fp32 -> bf16, row-major with padded stride
  for (int e = tid; e < ROWS_PER_WG * D_IN; e += 256) {
    int r = e >> 10, c = e & (D_IN - 1);
    float v = x[(size_t)(rowbase + r) * D_IN + c] - pre_b[c];
    As[r * AS_STRIDE + c] = (uint16_t)f32_to_bf16_bits(v);
  }

  // ---- register-staged enc pipeline: thread tid owns Es column n = tid.
  //      Loads for chunk kc+1 are issued before the WMMA loop of chunk kc,
  //      so global latency hides under 64 WMMAs.
  float sreg[KCHUNK];
  auto issue_loads = [&](int kc) {
    const float* p = enc + (size_t)kc * KCHUNK * N_FEAT + colbase + tid;
    #pragma unroll
    for (int k = 0; k < KCHUNK; ++k)
      sreg[k] = p[(size_t)k * N_FEAT];   // warp-coalesced (contiguous across tid)
  };
  auto store_chunk = [&]() {
    uint32_t* ep = (uint32_t*)&Es[tid * ES_STRIDE];
    #pragma unroll
    for (int j = 0; j < KCHUNK / 2; ++j) {
      uint32_t lo = f32_to_bf16_bits(sreg[2 * j]);
      uint32_t hi = f32_to_bf16_bits(sreg[2 * j + 1]);
      ep[j] = lo | (hi << 16);           // packed bf16 pair, stride-17-dword column
    }
  };

  v8f acc[8] = {};   // 8 accumulator tiles (16x16 f32) per wave
  issue_loads(0);

  for (int kc = 0; kc < NKC; ++kc) {
    __syncthreads();                     // previous chunk's consumers done
    store_chunk();                       // regs (chunk kc) -> Es
    __syncthreads();                     // Es ready
    if (kc + 1 < NKC) issue_loads(kc + 1);  // overlaps with WMMAs below

    // ---- A fragment (16x32 bf16) per the ISA 16-bit A layout
    union BF { v16bf v; uint32_t u[8]; };
    BF afrag;
    {
      const int rowA = sr * 16 + (lane & 15);
      const int hi   = (lane >= 16) ? 8 : 0;
      const uint16_t* ap = &As[rowA * AS_STRIDE + kc * KCHUNK];
      #pragma unroll
      for (int vv = 0; vv < 8; ++vv) {
        int kb = ((vv < 4) ? (2 * vv) : (16 + 2 * (vv - 4))) + hi;
        afrag.u[vv] = *(const uint32_t*)(ap + kb);   // aligned b32 pair (K, K+1)
      }
    }

    // ---- 8 WMMA tiles, B fragments double-buffered so ds_loads of tile
    //      t+1 overlap the WMMA of tile t
    auto loadB = [&](BF& f, int t) {
      const int n = g * 128 + t * 16 + (lane & 15);
      const uint16_t* bp = &Es[n * ES_STRIDE + ((lane >= 16) ? 8 : 0)];
      #pragma unroll
      for (int vv = 0; vv < 8; ++vv)
        f.u[vv] = *(const uint32_t*)(bp + ((vv < 4) ? (2 * vv) : (16 + 2 * (vv - 4))));   // aligned b32 pair
    };
    BF bf0, bf1;
    loadB(bf0, 0);
    #pragma unroll
    for (int t = 0; t < 8; ++t) {
      BF& cur = (t & 1) ? bf1 : bf0;
      BF& nxt = (t & 1) ? bf0 : bf1;
      if (t + 1 < 8) loadB(nxt, t + 1);
      acc[t] = __builtin_amdgcn_wmma_f32_16x16x32_bf16(
                   false, afrag.v, false, cur.v,
                   (short)0, acc[t], false, false);
    }
  }

  __syncthreads();
  // ---- init wave-local top-k buffers (alias over dead A tile)
  for (int e = tid; e < 8 * 16 * 32; e += 256) {
    twv[e] = -__builtin_inff();
    twi[e] = -1;
  }
  __syncthreads();

  // ---- streaming top-32 per (wave, row): min-replacement buffer
  float minv = -__builtin_inff();
  int   minslot = 0;
  #pragma unroll
  for (int t = 0; t < 8; ++t) {
    #pragma unroll
    for (int i = 0; i < 8; ++i) {       // C layout: VGPR i -> M = i (+8 for hi lanes)
      int r = (lane < 16) ? i : (8 + i);
      stageV[(w * 16 + r) * 16 + (lane & 15)] = acc[t][i];
    }
    __syncthreads();
    if (lane < 16) {                    // lane owns row rr of this wave's subtile
      const int rr = lane;
      float* vbuf = &twv[(w * 16 + rr) * 32];
      int*   ibuf = &twi[(w * 16 + rr) * 32];
      for (int c = 0; c < 16; ++c) {
        float v = stageV[(w * 16 + rr) * 16 + c];
        if (v > minv) {
          vbuf[minslot] = v;
          ibuf[minslot] = colbase + g * 128 + t * 16 + c;
          float mv = vbuf[0]; int ms = 0;
          for (int s = 1; s < 32; ++s) {
            float xv = vbuf[s];
            if (xv < mv) { mv = xv; ms = s; }
          }
          minv = mv; minslot = ms;
        }
      }
    }
    __syncthreads();
  }

  // ---- merge the 2 col-group buffers per row -> 32 candidates to workspace
  if (tid < ROWS_PER_WG) {
    const int r = tid, srr = r >> 4, rr = r & 15;
    const size_t outb = ((size_t)(rowbase + r) * NSLICE + slice) * TOPK;
    for (int s = 0; s < TOPK; ++s) {
      float best = -__builtin_inff(); int bw = 0, bs = 0;
      for (int gg = 0; gg < 2; ++gg)
        for (int j = 0; j < 32; ++j) {
          float v = twv[((srr * 2 + gg) * 16 + rr) * 32 + j];
          if (v > best) { best = v; bw = gg; bs = j; }
        }
      candV[outb + s] = best;
      candI[outb + s] = twi[((srr * 2 + bw) * 16 + rr) * 32 + bs];
      twv[((srr * 2 + bw) * 16 + rr) * 32 + bs] = -__builtin_inff();
    }
  }
}

// =====================================================================
// Kernel 2: merge 256 slices x 32 candidates -> final top-32 per row
// =====================================================================
__global__ __launch_bounds__(256)
void sae_merge_topk(const float* __restrict__ candV,
                    const int*   __restrict__ candI,
                    float* __restrict__ finV,
                    int*   __restrict__ finI)
{
  __shared__ float sv[NSLICE * TOPK];   // 32 KB
  __shared__ int   si[NSLICE * TOPK];   // 32 KB
  __shared__ float rv[256];
  __shared__ int   ri[256];
  const int tid = threadIdx.x;
  const int row = blockIdx.x;
  const size_t base = (size_t)row * NSLICE * TOPK;

  for (int e = tid; e < NSLICE * TOPK; e += 256) {
    sv[e] = candV[base + e];
    si[e] = candI[base + e];
  }
  __syncthreads();

  for (int s = 0; s < TOPK; ++s) {
    float best = -__builtin_inff(); int bi = 0;
    for (int e = tid; e < NSLICE * TOPK; e += 256) {
      float v = sv[e];
      if (v > best) { best = v; bi = e; }
    }
    rv[tid] = best; ri[tid] = bi;
    __syncthreads();
    for (int off = 128; off > 0; off >>= 1) {
      if (tid < off && rv[tid + off] > rv[tid]) {
        rv[tid] = rv[tid + off]; ri[tid] = ri[tid + off];
      }
      __syncthreads();
    }
    if (tid == 0) {
      finV[row * TOPK + s] = rv[0];
      finI[row * TOPK + s] = si[ri[0]];
      sv[ri[0]] = -__builtin_inff();
    }
    __syncthreads();
  }
}

// =====================================================================
// Kernel 3: sparse decode  out[row] = pre_b + sum_j val_j * dec[idx_j]
// =====================================================================
__global__ __launch_bounds__(256)
void sae_decode(const float* __restrict__ finV,
                const int*   __restrict__ finI,
                const float* __restrict__ dec,
                const float* __restrict__ pre_b,
                float* __restrict__ out)
{
  __shared__ float lv[TOPK];
  __shared__ int   li[TOPK];
  const int tid = threadIdx.x;
  const int row = blockIdx.x;
  if (tid < TOPK) { lv[tid] = finV[row * TOPK + tid]; li[tid] = finI[row * TOPK + tid]; }
  __syncthreads();

  const float4* pb4 = (const float4*)pre_b;
  float4 a = pb4[tid];                               // 256 threads x float4 = 1024
  #pragma unroll 8
  for (int j = 0; j < TOPK; ++j) {
    const float4* dr = (const float4*)(dec + (size_t)li[j] * D_IN);
    float4 d = dr[tid];
    float  v = lv[j];
    a.x += v * d.x; a.y += v * d.y; a.z += v * d.z; a.w += v * d.w;
  }
  ((float4*)out)[(size_t)row * (D_IN / 4) + tid] = a;
}

// =====================================================================
extern "C" void kernel_launch(void* const* d_in, const int* in_sizes, int n_in,
                              void* d_out, int out_size, void* d_ws, size_t ws_size,
                              hipStream_t stream)
{
  const float* x     = (const float*)d_in[0];
  const float* pre_b = (const float*)d_in[1];
  const float* enc   = (const float*)d_in[2];
  const float* dec   = (const float*)d_in[3];
  // d_in[4] is k (device scalar); K==32 is a compile-time constant here.
  (void)in_sizes; (void)n_in; (void)out_size; (void)ws_size;

  char* ws = (char*)d_ws;
  const size_t candElems = (size_t)B_ROWS * NSLICE * TOPK;   // 16M entries
  float* candV = (float*)ws;
  int*   candI = (int*)  (ws + candElems * sizeof(float));
  float* finV  = (float*)(ws + candElems * 8);
  int*   finI  = (int*)  (ws + candElems * 8 + (size_t)B_ROWS * TOPK * sizeof(float));

  dim3 gA(NROWT, NSLICE);   // row-tile is the FAST grid dim -> enc slices stay hot in L2
  sae_encode_topk<<<gA, 256, 0, stream>>>(x, pre_b, enc, candV, candI);
  sae_merge_topk<<<B_ROWS, 256, 0, stream>>>(candV, candI, finV, finI);
  sae_decode<<<B_ROWS, 256, 0, stream>>>(finV, finI, dec, pre_b, (float*)d_out);
}
